// TinyLLMAttention_11888469475735
// MI455X (gfx1250) — matrix operand order repaired and checked by reading, not compiled, not run on hardware
//
#include <hip/hip_runtime.h>

#define BSZ 2
#define SEQ 2048
#define DIM 2048
#define NH 16
#define HDIM 128

typedef __attribute__((ext_vector_type(16))) __bf16 bf16x16;
typedef __attribute__((ext_vector_type(8)))  float  f32x8;
typedef int v4i __attribute__((vector_size(16)));
typedef __attribute__((address_space(1))) v4i* gv4i_p; // global v4i*
typedef __attribute__((address_space(3))) v4i* lv4i_p; // LDS v4i*

union Frag { bf16x16 v; uint4 u[2]; };

// ---- CDNA5 async global->LDS copy (ASYNCcnt path), with safe fallback ------
#if defined(__has_builtin)
#  if __has_builtin(__builtin_amdgcn_global_load_async_to_lds_b128)
#    define HAVE_ASYNC_LDS 1
#  endif
#  if __has_builtin(__builtin_amdgcn_s_wait_asynccnt)
#    define HAVE_WAIT_ASYNC_BUILTIN 1
#  endif
#endif
#ifndef HAVE_ASYNC_LDS
#  define HAVE_ASYNC_LDS 0
#endif
#ifndef HAVE_WAIT_ASYNC_BUILTIN
#  define HAVE_WAIT_ASYNC_BUILTIN 0
#endif

__device__ __forceinline__ void cp16(unsigned short* l, const unsigned short* g) {
#if HAVE_ASYNC_LDS
  __builtin_amdgcn_global_load_async_to_lds_b128(
      (gv4i_p)(void*)g, (lv4i_p)(void*)l, 0, 0);
#else
  *(uint4*)l = *(const uint4*)g;
#endif
}
__device__ __forceinline__ void async_join() {
#if HAVE_ASYNC_LDS
#  if HAVE_WAIT_ASYNC_BUILTIN
  __builtin_amdgcn_s_wait_asynccnt(0);
#  else
  asm volatile("s_wait_asynccnt 0x0" ::: "memory");
#  endif
#endif
}

__device__ __forceinline__ f32x8 wmma_bf16(const Frag& a, const Frag& b, f32x8 c) {
  // D = A(16x32 bf16) x B(32x16 bf16) + C(16x16 f32)
  return __builtin_amdgcn_wmma_f32_16x16x32_bf16(false, a.v, false, b.v,
                                                 (short)0, c, false, false);
}

__device__ __forceinline__ unsigned short f2bf(float f) {
  union { float f; unsigned int u; } x; x.f = f;
  unsigned int u = x.u;
  return (unsigned short)((u + 0x7FFFu + ((u >> 16) & 1u)) >> 16); // RNE
}
__device__ __forceinline__ float bf2f(unsigned short h) {
  union { unsigned int u; float f; } x; x.u = ((unsigned int)h) << 16;
  return x.f;
}

__device__ __forceinline__ float redmax16(float v) {
  v = fmaxf(v, __shfl_xor(v, 1));
  v = fmaxf(v, __shfl_xor(v, 2));
  v = fmaxf(v, __shfl_xor(v, 4));
  v = fmaxf(v, __shfl_xor(v, 8));
  return v;
}
__device__ __forceinline__ float redsum16(float v) {
  v += __shfl_xor(v, 1);
  v += __shfl_xor(v, 2);
  v += __shfl_xor(v, 4);
  v += __shfl_xor(v, 8);
  return v;
}

// ---------------- fp32 -> bf16 conversion (2 elems / thread) ----------------
__global__ __launch_bounds__(256)
void cvt_kernel(const float* __restrict__ src, unsigned int* __restrict__ dst, int n2) {
  int i = blockIdx.x * blockDim.x + threadIdx.x;
  if (i < n2) {
    float a = src[2 * i], b = src[2 * i + 1];
    dst[i] = (unsigned int)f2bf(a) | ((unsigned int)f2bf(b) << 16);
  }
}

// ---------------- RoPE in place on bf16 Q (with 1/sqrt(HD) fold) and K -------
__global__ __launch_bounds__(256)
void rope_kernel(unsigned int* __restrict__ Q, unsigned int* __restrict__ K,
                 const float* __restrict__ C, const float* __restrict__ Sn) {
  int idx = blockIdx.x * blockDim.x + threadIdx.x; // < B*S*H*64 = 2^22
  int i = idx & 63;
  int h = (idx >> 6) & (NH - 1);
  int s = (idx >> 10) & (SEQ - 1);
  int b = (idx >> 21) & 1;
  float c  = C[s * 64 + i];
  float sn = Sn[s * 64 + i];
  size_t w = (size_t)(b * SEQ + s) * (DIM / 2) + h * (HDIM / 2) + i;
  const float qscale = 0.08838834764831845f; // 1/sqrt(128)
  {
    unsigned int u = Q[w];
    float xr = bf2f((unsigned short)(u & 0xffffu));
    float xi = bf2f((unsigned short)(u >> 16));
    float orr = (xr * c - xi * sn) * qscale;
    float oii = (xr * sn + xi * c) * qscale;
    Q[w] = (unsigned int)f2bf(orr) | ((unsigned int)f2bf(oii) << 16);
  }
  {
    unsigned int u = K[w];
    float xr = bf2f((unsigned short)(u & 0xffffu));
    float xi = bf2f((unsigned short)(u >> 16));
    float orr = xr * c - xi * sn;
    float oii = xr * sn + xi * c;
    K[w] = (unsigned int)f2bf(orr) | ((unsigned int)f2bf(oii) << 16);
  }
}

// ---------------- C = A(MxK) * W(NxK)^T, bf16 in, f32 acc --------------------
// Block tile 128x128, 8 waves (2M x 4N), wave tile 64x32, K-step 64 via LDS.
template <bool BF16OUT>
__global__ __launch_bounds__(256)
void gemm_tn(const unsigned short* __restrict__ A, const unsigned short* __restrict__ W,
             void* __restrict__ Cout) {
  __shared__ unsigned short As[128 * 72]; // stride 72 halves = 144B (16B multiple)
  __shared__ unsigned short Bs[128 * 72];
  const int tid = threadIdx.x;
  const int wid = tid >> 5, lane = tid & 31;
  const int l16 = lane & 15, hiw = lane >> 4;
  const int waveM = wid & 1, waveN = wid >> 1;
  const int mBlock = blockIdx.y * 128, nBlock = blockIdx.x * 128;
  const int kb = hiw * 8;

  f32x8 acc[4][2] = {};

  const int ldr = tid >> 1, ldc = (tid & 1) * 32;
  const unsigned short* Ag = A + (size_t)(mBlock + ldr) * DIM + ldc;
  const unsigned short* Wg = W + (size_t)(nBlock + ldr) * DIM + ldc;
  unsigned short* asr = As + ldr * 72 + ldc;
  unsigned short* bsr = Bs + ldr * 72 + ldc;

  for (int k0 = 0; k0 < DIM; k0 += 64) {
    __syncthreads();
#pragma unroll
    for (int j = 0; j < 4; ++j) {
      cp16(asr + j * 8, Ag + k0 + j * 8);
      cp16(bsr + j * 8, Wg + k0 + j * 8);
    }
    if (k0 + 64 < DIM) {
      __builtin_prefetch(Ag + k0 + 64, 0, 1); // global_prefetch_b8
      __builtin_prefetch(Wg + k0 + 64, 0, 1);
    }
    async_join();
    __syncthreads();

#pragma unroll
    for (int ks = 0; ks < 2; ++ks) {
      Frag a[4], bfr[2];
#pragma unroll
      for (int mi = 0; mi < 4; ++mi) {
        // A 16x32 bf16 lane layout: lane<16 -> K {0..7,16..23}, lane>=16 -> +8
        const unsigned short* p = As + (waveM * 64 + mi * 16 + l16) * 72 + ks * 32 + kb;
        a[mi].u[0] = *(const uint4*)(p);
        a[mi].u[1] = *(const uint4*)(p + 16);
      }
#pragma unroll
      for (int ni = 0; ni < 2; ++ni) {
        // B 32x16 bf16 lane layout: N=lane%16, lane<16 -> K {0..7,16..23}, lane>=16 -> +8
        const unsigned short* p = Bs + (waveN * 32 + ni * 16 + l16) * 72 + ks * 32 + kb;
        bfr[ni].u[0] = *(const uint4*)(p);
        bfr[ni].u[1] = *(const uint4*)(p + 16);
      }
#pragma unroll
      for (int mi = 0; mi < 4; ++mi)
#pragma unroll
        for (int ni = 0; ni < 2; ++ni)
          acc[mi][ni] = wmma_bf16(a[mi], bfr[ni], acc[mi][ni]);
    }
  }

#pragma unroll
  for (int mi = 0; mi < 4; ++mi)
#pragma unroll
    for (int ni = 0; ni < 2; ++ni)
#pragma unroll
      for (int r = 0; r < 8; ++r) {
        int row = mBlock + waveM * 64 + mi * 16 + r + 8 * hiw;
        int col = nBlock + waveN * 32 + ni * 16 + l16;
        float v = acc[mi][ni][r];
        if (BF16OUT)
          ((unsigned short*)Cout)[(size_t)row * DIM + col] = f2bf(v);
        else
          ((float*)Cout)[(size_t)row * DIM + col] = v;
      }
}

// ---------------- Flash attention: one WG per (b, head, 128 q-rows) ----------
// 64 keys per iteration: halves per-key softmax/rescale overhead vs 32.
__global__ __launch_bounds__(256)
void attn_kernel(const unsigned short* __restrict__ Qb,
                 const unsigned short* __restrict__ Kb,
                 const unsigned short* __restrict__ Vb,
                 unsigned short* __restrict__ Ob) {
  __shared__ unsigned short Ks[64 * 136]; // [key][d], stride 136 halves
  __shared__ unsigned short Vt[128 * 72]; // [d][key], transposed at load
  __shared__ unsigned short Ps[8 * 16 * 72]; // per-wave 16x64 P tile, stride 72

  const int qt = blockIdx.x, head = blockIdx.y, b = blockIdx.z;
  const int tid = threadIdx.x, wid = tid >> 5, lane = tid & 31;
  const int l16 = lane & 15, hiw = lane >> 4;
  const int kb = hiw * 8;
  const int q_base = qt * 128;
  const size_t base_bh = (size_t)b * SEQ * DIM + (size_t)head * HDIM;

  // Q fragments resident in registers (d slices 0..127 in steps of 32)
  Frag qf[4];
  {
    const int qrow = q_base + wid * 16 + l16;
    const unsigned short* qp = Qb + base_bh + (size_t)qrow * DIM;
#pragma unroll
    for (int dk = 0; dk < 4; ++dk) {
      qf[dk].u[0] = *(const uint4*)(qp + dk * 32 + kb);
      qf[dk].u[1] = *(const uint4*)(qp + dk * 32 + kb + 16);
    }
  }

  f32x8 o[8] = {};
  float mrow[8], lrow[8];
#pragma unroll
  for (int r = 0; r < 8; ++r) { mrow[r] = -1e30f; lrow[r] = 0.f; }

  const int ldr = tid >> 2;        // key row 0..63
  const int ldc = (tid & 3) * 32;  // d col 0/32/64/96
  unsigned short* Pw = Ps + wid * (16 * 72);

  const int nkb = qt * 2 + 2;      // causal: keys only up to q_base+127
  for (int jb = 0; jb < nkb; ++jb) {
    const int key0 = jb * 64;
    __syncthreads();
    {
      // K tile via async global->LDS DMA (no VGPR round-trip)
      const unsigned short* kg = Kb + base_bh + (size_t)(key0 + ldr) * DIM + ldc;
#pragma unroll
      for (int j = 0; j < 4; ++j)
        cp16(Ks + ldr * 136 + ldc + j * 8, kg + j * 8);
      // V tile transposed through VGPRs
      const unsigned short* vg = Vb + base_bh + (size_t)(key0 + ldr) * DIM + ldc;
      union { uint4 q; unsigned short s[8]; } vv[4];
#pragma unroll
      for (int j = 0; j < 4; ++j) vv[j].q = *(const uint4*)(vg + j * 8);
#pragma unroll
      for (int j = 0; j < 4; ++j)
#pragma unroll
        for (int i = 0; i < 8; ++i)
          Vt[(ldc + j * 8 + i) * 72 + ldr] = vv[j].s[i];
      if (jb + 1 < nkb) {
        __builtin_prefetch(kg + DIM * 64, 0, 1);
        __builtin_prefetch(vg + DIM * 64, 0, 1);
      }
    }
    async_join();
    __syncthreads();

    // S = Q * K^T : 16 q rows x 64 keys, contract over d=128 in steps of 32
    f32x8 s4[4] = {};
#pragma unroll
    for (int dk = 0; dk < 4; ++dk) {
#pragma unroll
      for (int a2 = 0; a2 < 4; ++a2) {
        Frag kf;
        const unsigned short* p = Ks + (a2 * 16 + l16) * 136 + dk * 32 + kb;
        kf.u[0] = *(const uint4*)(p);
        kf.u[1] = *(const uint4*)(p + 16);
        s4[a2] = wmma_bf16(qf[dk], kf, s4[a2]);
      }
    }

    if (key0 >= q_base) { // only diagonal blocks need the causal mask
#pragma unroll
      for (int r = 0; r < 8; ++r) {
        int qr = q_base + wid * 16 + r + 8 * hiw;
#pragma unroll
        for (int a2 = 0; a2 < 4; ++a2)
          if (key0 + a2 * 16 + l16 > qr) s4[a2][r] = -1e30f;
      }
    }

    // online softmax (Q already scaled by 1/sqrt(HD))
#pragma unroll
    for (int r = 0; r < 8; ++r) {
      float mx = fmaxf(fmaxf(s4[0][r], s4[1][r]), fmaxf(s4[2][r], s4[3][r]));
      mx = redmax16(mx);
      float mnew = fmaxf(mrow[r], mx);
      float alpha = __expf(mrow[r] - mnew);
      float p0 = __expf(s4[0][r] - mnew);
      float p1 = __expf(s4[1][r] - mnew);
      float p2 = __expf(s4[2][r] - mnew);
      float p3 = __expf(s4[3][r] - mnew);
      lrow[r] = lrow[r] * alpha + redsum16((p0 + p1) + (p2 + p3));
      mrow[r] = mnew;
#pragma unroll
      for (int t = 0; t < 8; ++t) o[t][r] *= alpha;
      int prow = r + 8 * hiw;
      Pw[prow * 72 + l16]      = f2bf(p0);
      Pw[prow * 72 + 16 + l16] = f2bf(p1);
      Pw[prow * 72 + 32 + l16] = f2bf(p2);
      Pw[prow * 72 + 48 + l16] = f2bf(p3);
    }

    asm volatile("s_wait_dscnt 0x0" ::: "memory"); // P stores -> P frag loads

    Frag pf[2];
#pragma unroll
    for (int ks = 0; ks < 2; ++ks) {
      const unsigned short* pp = Pw + l16 * 72 + ks * 32 + kb;
      pf[ks].u[0] = *(const uint4*)(pp);
      pf[ks].u[1] = *(const uint4*)(pp + 16);
    }

#pragma unroll
    for (int t = 0; t < 8; ++t) { // O += P(16x64) * V(64x16) per d-slice
#pragma unroll
      for (int ks = 0; ks < 2; ++ks) {
        Frag vf;
        const unsigned short* vp = Vt + (t * 16 + l16) * 72 + ks * 32 + kb;
        vf.u[0] = *(const uint4*)(vp);
        vf.u[1] = *(const uint4*)(vp + 16);
        o[t] = wmma_bf16(pf[ks], vf, o[t]);
      }
    }
  }

#pragma unroll
  for (int r = 0; r < 8; ++r) {
    float inv = 1.f / lrow[r];
    int row = q_base + wid * 16 + r + 8 * hiw;
    unsigned short* op = Ob + (size_t)(b * SEQ + row) * DIM + head * HDIM;
#pragma unroll
    for (int t = 0; t < 8; ++t)
      op[t * 16 + l16] = f2bf(o[t][r] * inv);
  }
}

// ---------------- host side -------------------------------------------------
extern "C" void kernel_launch(void* const* d_in, const int* in_sizes, int n_in,
                              void* d_out, int out_size, void* d_ws, size_t ws_size,
                              hipStream_t stream) {
  (void)in_sizes; (void)n_in; (void)out_size; (void)ws_size;
  const float* X  = (const float*)d_in[0];
  const float* fc = (const float*)d_in[1];
  const float* fs = (const float*)d_in[2];
  // d_in[3] = attention_mask: unused, causal mask is computed analytically
  const float* wq = (const float*)d_in[4];
  const float* wk = (const float*)d_in[5];
  const float* wv = (const float*)d_in[6];
  const float* wo = (const float*)d_in[7];

  char* w = (char*)d_ws; // 112 MB used, all offsets 16B-aligned
  unsigned short* Xb  = (unsigned short*)(w + 0);          // 16 MB
  unsigned short* Wqb = (unsigned short*)(w + 16777216);   //  8 MB
  unsigned short* Wkb = (unsigned short*)(w + 25165824);
  unsigned short* Wvb = (unsigned short*)(w + 33554432);
  unsigned short* Wob = (unsigned short*)(w + 41943040);
  unsigned short* Qb  = (unsigned short*)(w + 50331648);   // 16 MB each
  unsigned short* Kb  = (unsigned short*)(w + 67108864);
  unsigned short* Vb  = (unsigned short*)(w + 83886080);
  unsigned short* Ab  = (unsigned short*)(w + 100663296);

  const int nX = BSZ * SEQ * DIM; // 8,388,608
  const int nW = DIM * DIM;       // 4,194,304
  cvt_kernel<<<nX / 512, 256, 0, stream>>>(X,  (unsigned int*)Xb,  nX / 2);
  cvt_kernel<<<nW / 512, 256, 0, stream>>>(wq, (unsigned int*)Wqb, nW / 2);
  cvt_kernel<<<nW / 512, 256, 0, stream>>>(wk, (unsigned int*)Wkb, nW / 2);
  cvt_kernel<<<nW / 512, 256, 0, stream>>>(wv, (unsigned int*)Wvb, nW / 2);
  cvt_kernel<<<nW / 512, 256, 0, stream>>>(wo, (unsigned int*)Wob, nW / 2);

  dim3 gg(DIM / 128, (BSZ * SEQ) / 128);
  gemm_tn<true><<<gg, 256, 0, stream>>>(Xb, Wqb, Qb);
  gemm_tn<true><<<gg, 256, 0, stream>>>(Xb, Wkb, Kb);
  gemm_tn<true><<<gg, 256, 0, stream>>>(Xb, Wvb, Vb);

  rope_kernel<<<(BSZ * SEQ * NH * 64) / 256, 256, 0, stream>>>(
      (unsigned int*)Qb, (unsigned int*)Kb, fc, fs);

  dim3 ga(SEQ / 128, NH, BSZ);
  attn_kernel<<<ga, 256, 0, stream>>>(Qb, Kb, Vb, Ab);

  gemm_tn<false><<<gg, 256, 0, stream>>>(Ab, Wob, d_out);
}
